// DecoderLayer_31026843746997
// MI455X (gfx1250) — matrix operand order repaired and checked by reading, not compiled, not run on hardware
//
#include <hip/hip_runtime.h>

// ---------------------------------------------------------------------------
// Transformer decoder layer for MI455X (gfx1250), bf16 WMMA everywhere.
// B=4, S=1024, D=1024, H=16, DK=64, DFF=4096. ~172 GFLOP -> matrix-core bound.
// All WMMA operands are stored contiguously along the K-dimension (weights
// pre-transposed, V pre-transposed per head) so every fragment load is two
// 16-byte b128 loads. GEMM uses CDNA5 async global->LDS copies (ASYNCcnt)
// with double-buffered LDS when the toolchain exposes the builtin.
// ---------------------------------------------------------------------------

#define B_   4
#define S_   1024
#define D_   1024
#define H_   16
#define DK_  64
#define DFF_ 4096
#define NTOK (B_ * S_)

typedef unsigned short u16;
typedef unsigned int   u32;
typedef __attribute__((ext_vector_type(16))) __bf16 v16bf;
typedef __attribute__((ext_vector_type(8)))  float  v8f;

union frag16 { uint4 q[2]; v16bf v; };   // 32B fragment = two b128 chunks

__device__ __forceinline__ u16 f2bf(float f) {
  u32 u = __float_as_uint(f);
  u32 r = u + 0x7FFFu + ((u >> 16) & 1u);   // round-to-nearest-even
  return (u16)(r >> 16);
}

// ---- CDNA5 async global->LDS copy (bypasses VGPRs, tracked by ASYNCcnt) ----
#if __has_builtin(__builtin_amdgcn_global_load_async_to_lds_b128)
#define HAVE_ASYNC_COPY 1
typedef int i4v __attribute__((ext_vector_type(4)));
typedef __attribute__((address_space(1))) void* gasv_t;
typedef __attribute__((address_space(3))) void* lasv_t;
typedef __attribute__((address_space(1))) i4v*  gas1_t;
typedef __attribute__((address_space(3))) i4v*  las3_t;
__device__ __forceinline__ void async_ld_b128(const void* g, void* l) {
  __builtin_amdgcn_global_load_async_to_lds_b128(
      (gas1_t)(gasv_t)const_cast<void*>(g), (las3_t)(lasv_t)l, 0, 0);
}
__device__ __forceinline__ void wait_async_3() {
#if __has_builtin(__builtin_amdgcn_s_wait_asynccnt)
  __builtin_amdgcn_s_wait_asynccnt(3);
#else
  asm volatile("s_wait_asynccnt 3" ::: "memory");
#endif
}
__device__ __forceinline__ void wait_async_0() {
#if __has_builtin(__builtin_amdgcn_s_wait_asynccnt)
  __builtin_amdgcn_s_wait_asynccnt(0);
#else
  asm volatile("s_wait_asynccnt 0" ::: "memory");
#endif
}
#endif

// ---------------------------------------------------------------------------
// fp32 -> bf16 elementwise convert (encoder activations)
// ---------------------------------------------------------------------------
__global__ void cvt_f32_bf16(const float* __restrict__ in, u16* __restrict__ out, int n) {
  int i = blockIdx.x * blockDim.x + threadIdx.x;
  int stride = gridDim.x * blockDim.x;
  for (; i < n; i += stride) out[i] = f2bf(in[i]);
}

// ---------------------------------------------------------------------------
// fp32 W[K][N] -> bf16 Wt[N][K] tiled transpose-convert (32x32 tiles)
// block (32,8)
// ---------------------------------------------------------------------------
__global__ __launch_bounds__(256)
void transpose_cvt(const float* __restrict__ W, u16* __restrict__ Wt, int K, int N) {
  __shared__ float tile[32][33];
  const int nb = blockIdx.x * 32, kb = blockIdx.y * 32;
  const int tx = threadIdx.x, ty = threadIdx.y;
  #pragma unroll
  for (int i = 0; i < 32; i += 8)
    tile[ty + i][tx] = W[(size_t)(kb + ty + i) * N + nb + tx];   // tile[k][n]
  __syncthreads();
  #pragma unroll
  for (int i = 0; i < 32; i += 8)
    Wt[(size_t)(nb + ty + i) * K + kb + tx] = f2bf(tile[tx][ty + i]);
}

// ---------------------------------------------------------------------------
// bf16 V[b*S+s][D] (head h at cols h*64..) -> Vt[(b*H+h)*DK+d][S] per-head
// transpose. block (32,8), grid (S/32, DK/32, B*H)
// ---------------------------------------------------------------------------
__global__ __launch_bounds__(256)
void transpose_head(const u16* __restrict__ V, u16* __restrict__ Vt) {
  __shared__ u16 tile[32][33];
  const int bh = blockIdx.z, b = bh / H_, h = bh % H_;
  const int sb = blockIdx.x * 32, db = blockIdx.y * 32;
  const int tx = threadIdx.x, ty = threadIdx.y;
  #pragma unroll
  for (int i = 0; i < 32; i += 8)
    tile[ty + i][tx] = V[(size_t)(b * S_ + sb + ty + i) * D_ + h * DK_ + db + tx];
  __syncthreads();
  #pragma unroll
  for (int i = 0; i < 32; i += 8)
    Vt[((size_t)bh * DK_ + db + ty + i) * S_ + sb + tx] = tile[tx][ty + i];
}

// ---------------------------------------------------------------------------
// LayerNorm over D=1024, one row per 256-thread block, bf16 output
// ---------------------------------------------------------------------------
__global__ __launch_bounds__(256)
void layernorm_bf16(const float* __restrict__ X, const float* __restrict__ g,
                    const float* __restrict__ bta, u16* __restrict__ Out) {
  __shared__ float s1[256], s2[256];
  const int row = blockIdx.x, tid = threadIdx.x;
  const float4 xv = *(const float4*)(X + (size_t)row * D_ + tid * 4);
  s1[tid] = xv.x + xv.y + xv.z + xv.w;
  s2[tid] = xv.x * xv.x + xv.y * xv.y + xv.z * xv.z + xv.w * xv.w;
  __syncthreads();
  for (int st = 128; st > 0; st >>= 1) {
    if (tid < st) { s1[tid] += s1[tid + st]; s2[tid] += s2[tid + st]; }
    __syncthreads();
  }
  const float mu   = s1[0] * (1.0f / D_);
  const float var  = s2[0] * (1.0f / D_) - mu * mu;
  const float rstd = rsqrtf(var + 1e-5f);
  const float4 gv = *(const float4*)(g   + tid * 4);
  const float4 bv = *(const float4*)(bta + tid * 4);
  u16* op = Out + (size_t)row * D_ + tid * 4;
  op[0] = f2bf((xv.x - mu) * rstd * gv.x + bv.x);
  op[1] = f2bf((xv.y - mu) * rstd * gv.y + bv.y);
  op[2] = f2bf((xv.z - mu) * rstd * gv.z + bv.z);
  op[3] = f2bf((xv.w - mu) * rstd * gv.w + bv.w);
}

// ---------------------------------------------------------------------------
// Tiled bf16 WMMA GEMM: C[M,N] = A[M,K] * Wt[N,K]^T + bias
//   A row-major [M,K]; weights pre-transposed to [N,K] (K contiguous).
//   mode 0: store bf16   mode 1: store bf16 after exact GELU
//   mode 2: Cf[M,N] += result (fp32 residual accumulate)
// Block = 256 threads = 8 waves; BM=128, BN=64, BK=32; 2x2 frags per wave.
// LDS rows padded to 96B so all fragment loads are ds_load_b128.
// With async copies: double-buffered LDS, tile t+1 DMA overlaps tile t WMMAs.
// ---------------------------------------------------------------------------
#define BM 128
#define BN 64
#define BK 32
#define LDP 48   // padded row length (u16) -> 96B, multiple of 16B

__global__ __launch_bounds__(256)
void gemm_bf16_wmma(const u16* __restrict__ A, const u16* __restrict__ Wt,
                    const float* __restrict__ bias,
                    u16* __restrict__ Cb, float* __restrict__ Cf,
                    int M, int N, int K, int mode) {
  const int tid  = threadIdx.x;
  const int lane = tid & 31, wid = tid >> 5;
  const int waveM = wid >> 1, waveN = wid & 1;     // 4x2 wave grid
  const int mBase = blockIdx.y * BM;
  const int nBase = blockIdx.x * BN;
  const int lg = lane & 15, half = lane >> 4;

  v8f acc[2][2];
  #pragma unroll
  for (int i = 0; i < 2; i++)
    #pragma unroll
    for (int j = 0; j < 2; j++)
      #pragma unroll
      for (int r = 0; r < 8; r++) acc[i][j][r] = 0.0f;

  // static staging assignments: A tile = 512 b128 chunks, B tile = 256
  const int ar0 = tid >> 2,          aq0 = tid & 3;          // chunks 0..255
  const int ar1 = (tid + 256) >> 2,  aq1 = tid & 3;          // chunks 256..511
  const int br  = tid >> 2,          bq  = tid & 3;          // 64 rows x 4

#ifdef HAVE_ASYNC_COPY
  // ---- CDNA5 async double-buffered pipeline ----
  __shared__ u16 As[2][BM][LDP];
  __shared__ u16 Bs[2][BN][LDP];
  const int T = K / BK;
  {
    // prologue: tile 0 -> buffer 0
    async_ld_b128(A  + (size_t)(mBase + ar0) * K + aq0 * 8, &As[0][ar0][aq0 * 8]);
    async_ld_b128(A  + (size_t)(mBase + ar1) * K + aq1 * 8, &As[0][ar1][aq1 * 8]);
    async_ld_b128(Wt + (size_t)(nBase + br)  * K + bq  * 8, &Bs[0][br][bq * 8]);
  }
  for (int t = 0; t < T; t++) {
    const int cur = t & 1;
    const int kk1 = (t + 1) * BK;
    if (t + 1 < T) {   // prefetch tile t+1 into the other buffer via DMA
      async_ld_b128(A  + (size_t)(mBase + ar0) * K + kk1 + aq0 * 8, &As[1 - cur][ar0][aq0 * 8]);
      async_ld_b128(A  + (size_t)(mBase + ar1) * K + kk1 + aq1 * 8, &As[1 - cur][ar1][aq1 * 8]);
      async_ld_b128(Wt + (size_t)(nBase + br)  * K + kk1 + bq  * 8, &Bs[1 - cur][br][bq * 8]);
      wait_async_3();   // in-order completion => tile t's 3 loads have landed
    } else {
      wait_async_0();
    }
    __syncthreads();

    frag16 afr[2], bfr[2];
    #pragma unroll
    for (int mi = 0; mi < 2; mi++) {
      const int row = waveM * 32 + mi * 16 + lg;
      afr[mi].q[0] = *(const uint4*)&As[cur][row][8 * half];
      afr[mi].q[1] = *(const uint4*)&As[cur][row][16 + 8 * half];
    }
    #pragma unroll
    for (int ni = 0; ni < 2; ni++) {
      const int col = waveN * 32 + ni * 16 + lg;
      bfr[ni].q[0] = *(const uint4*)&Bs[cur][col][8 * half];
      bfr[ni].q[1] = *(const uint4*)&Bs[cur][col][16 + 8 * half];
    }
    #pragma unroll
    for (int mi = 0; mi < 2; mi++)
      #pragma unroll
      for (int ni = 0; ni < 2; ni++)
        acc[mi][ni] = __builtin_amdgcn_wmma_f32_16x16x32_bf16(
            false, afr[mi].v, false, bfr[ni].v, (short)0, acc[mi][ni], false, false);
    __syncthreads();   // all waves done reading buffer `cur` before its refill
  }
#else
  // ---- synchronous fallback (VGPR round trip + global_prefetch) ----
  __shared__ u16 As[BM][LDP];
  __shared__ u16 Bs[BN][LDP];
  for (int kk = 0; kk < K; kk += BK) {
    const uint4 va0 = *(const uint4*)(A  + (size_t)(mBase + ar0) * K + kk + aq0 * 8);
    const uint4 va1 = *(const uint4*)(A  + (size_t)(mBase + ar1) * K + kk + aq1 * 8);
    const uint4 vb0 = *(const uint4*)(Wt + (size_t)(nBase + br)  * K + kk + bq  * 8);
    if (kk + BK < K) {
      __builtin_prefetch(A  + (size_t)(mBase + ar0) * K + kk + BK, 0, 1);
      __builtin_prefetch(Wt + (size_t)(nBase + br)  * K + kk + BK, 0, 1);
    }
    *(uint4*)&As[ar0][aq0 * 8] = va0;
    *(uint4*)&As[ar1][aq1 * 8] = va1;
    *(uint4*)&Bs[br][bq * 8]   = vb0;
    __syncthreads();

    frag16 afr[2], bfr[2];
    #pragma unroll
    for (int mi = 0; mi < 2; mi++) {
      const int row = waveM * 32 + mi * 16 + lg;
      afr[mi].q[0] = *(const uint4*)&As[row][8 * half];
      afr[mi].q[1] = *(const uint4*)&As[row][16 + 8 * half];
    }
    #pragma unroll
    for (int ni = 0; ni < 2; ni++) {
      const int col = waveN * 32 + ni * 16 + lg;
      bfr[ni].q[0] = *(const uint4*)&Bs[col][8 * half];
      bfr[ni].q[1] = *(const uint4*)&Bs[col][16 + 8 * half];
    }
    #pragma unroll
    for (int mi = 0; mi < 2; mi++)
      #pragma unroll
      for (int ni = 0; ni < 2; ni++)
        acc[mi][ni] = __builtin_amdgcn_wmma_f32_16x16x32_bf16(
            false, afr[mi].v, false, bfr[ni].v, (short)0, acc[mi][ni], false, false);
    __syncthreads();
  }
#endif

  // epilogue: C layout -> VGPR r holds row r + 8*half, col = lg across lanes
  #pragma unroll
  for (int mi = 0; mi < 2; mi++)
    #pragma unroll
    for (int ni = 0; ni < 2; ni++) {
      const int col = nBase + waveN * 32 + ni * 16 + lg;
      const float bb = bias[col];
      #pragma unroll
      for (int r = 0; r < 8; r++) {
        const int row = mBase + waveM * 32 + mi * 16 + r + 8 * half;
        const float v = acc[mi][ni][r] + bb;
        if (mode == 0) {
          Cb[(size_t)row * N + col] = f2bf(v);
        } else if (mode == 1) {
          const float gl = 0.5f * v * (1.0f + erff(v * 0.70710678118f));
          Cb[(size_t)row * N + col] = f2bf(gl);
        } else {
          Cf[(size_t)row * N + col] += v;
        }
      }
    }
}

// ---------------------------------------------------------------------------
// Flash attention, bf16 WMMA, fp32 online softmax.
// Grid: (S/128, H, B); block 256 = 8 waves; each wave owns a 16-row Q strip.
// Q/K: [B*S, D] bf16, head h at cols h*64.. (K-run contiguous along head dim)
// Vt : [(b*H+h)*DK + d][S] bf16 (K-run contiguous along sequence)
// ---------------------------------------------------------------------------
__global__ __launch_bounds__(256)
void attn_wmma(const u16* __restrict__ Q, const u16* __restrict__ Kp,
               const u16* __restrict__ Vt, u16* __restrict__ Out, int causal) {
  __shared__ u16 Plds[8][16][LDP];   // per-wave 16x32 P tile (C->A relayout)
  const int tid = threadIdx.x, lane = tid & 31, wid = tid >> 5;
  const int lg = lane & 15, half = lane >> 4;
  const int h = blockIdx.y, b = blockIdx.z;
  const int qbase = blockIdx.x * 128 + wid * 16;
  const size_t tokBase = (size_t)b * S_;
  const int hcol = h * DK_;

  // Q fragments: 16x64 = two 16x32 A-frags (d 0..31 and 32..63)
  frag16 q0, q1;
  {
    const size_t ro = (tokBase + qbase + lg) * D_ + hcol;
    q0.q[0] = *(const uint4*)(Q + ro + 8 * half);
    q0.q[1] = *(const uint4*)(Q + ro + 16 + 8 * half);
    q1.q[0] = *(const uint4*)(Q + ro + 32 + 8 * half);
    q1.q[1] = *(const uint4*)(Q + ro + 48 + 8 * half);
  }

  float mrow[8], lrow[8];
  v8f o[4];
  #pragma unroll
  for (int r = 0; r < 8; r++) { mrow[r] = -3.0e38f; lrow[r] = 0.0f; }
  #pragma unroll
  for (int c = 0; c < 4; c++)
    #pragma unroll
    for (int r = 0; r < 8; r++) o[c][r] = 0.0f;

  const int ntiles = causal ? ((qbase + 47) >> 5) : (S_ >> 5);

  for (int kt = 0; kt < ntiles; kt++) {
    const int kb = kt * 32;
    v8f sf[2];
    #pragma unroll
    for (int nh = 0; nh < 2; nh++) {
      const int kcol = kb + nh * 16 + lg;
      const u16* kptr = Kp + (tokBase + kcol) * D_ + hcol;
      frag16 bk0, bk1;                    // d = 8*half+e, 16+8*half+e (+32 for bk1)
      bk0.q[0] = *(const uint4*)(kptr + 8 * half);
      bk0.q[1] = *(const uint4*)(kptr + 16 + 8 * half);
      bk1.q[0] = *(const uint4*)(kptr + 32 + 8 * half);
      bk1.q[1] = *(const uint4*)(kptr + 48 + 8 * half);
      v8f s;
      #pragma unroll
      for (int r = 0; r < 8; r++) s[r] = 0.0f;
      s = __builtin_amdgcn_wmma_f32_16x16x32_bf16(false, q0.v, false, bk0.v, (short)0, s, false, false);
      s = __builtin_amdgcn_wmma_f32_16x16x32_bf16(false, q1.v, false, bk1.v, (short)0, s, false, false);
      sf[nh] = s;
    }
    // scale 1/sqrt(64) + causal mask
    #pragma unroll
    for (int nh = 0; nh < 2; nh++) {
      const int kc = kb + nh * 16 + lg;
      #pragma unroll
      for (int r = 0; r < 8; r++) {
        float s = sf[nh][r] * 0.125f;
        if (causal && kc > qbase + r + 8 * half) s = -1.0e30f;
        sf[nh][r] = s;
      }
    }
    // online softmax (row = r + 8*half; 16-lane group reductions)
    #pragma unroll
    for (int r = 0; r < 8; r++) {
      float pm = fmaxf(sf[0][r], sf[1][r]);
      #pragma unroll
      for (int off = 8; off >= 1; off >>= 1) pm = fmaxf(pm, __shfl_xor(pm, off, 32));
      const float mnew  = fmaxf(mrow[r], pm);
      const float alpha = __expf(mrow[r] - mnew);
      const float p0 = __expf(sf[0][r] - mnew);
      const float p1 = __expf(sf[1][r] - mnew);
      float ps = p0 + p1;
      #pragma unroll
      for (int off = 8; off >= 1; off >>= 1) ps += __shfl_xor(ps, off, 32);
      lrow[r] = lrow[r] * alpha + ps;
      mrow[r] = mnew;
      #pragma unroll
      for (int c = 0; c < 4; c++) o[c][r] *= alpha;
      const int row = r + 8 * half;
      Plds[wid][row][lg]      = f2bf(p0);
      Plds[wid][row][16 + lg] = f2bf(p1);
    }
    // P as A-fragment (C layout -> A layout via per-wave LDS round trip)
    frag16 pf;
    pf.q[0] = *(const uint4*)&Plds[wid][lg][8 * half];
    pf.q[1] = *(const uint4*)&Plds[wid][lg][16 + 8 * half];
    // V B-fragments (contiguous along s in Vt) and P@V accumulation
    #pragma unroll
    for (int c = 0; c < 4; c++) {
      const size_t vb = ((size_t)(b * H_ + h) * DK_ + c * 16 + lg) * S_ + kb + 8 * half;
      frag16 bv;
      bv.q[0] = *(const uint4*)(Vt + vb);
      bv.q[1] = *(const uint4*)(Vt + vb + 16);
      o[c] = __builtin_amdgcn_wmma_f32_16x16x32_bf16(false, pf.v, false, bv.v, (short)0, o[c], false, false);
    }
  }

  #pragma unroll
  for (int r = 0; r < 8; r++) {
    const float inv = 1.0f / lrow[r];
    const size_t row = tokBase + qbase + r + 8 * half;
    #pragma unroll
    for (int c = 0; c < 4; c++)
      Out[row * D_ + hcol + c * 16 + lg] = f2bf(o[c][r] * inv);
  }
}

// ---------------------------------------------------------------------------
// Host-side orchestration
// ---------------------------------------------------------------------------
extern "C" void kernel_launch(void* const* d_in, const int* in_sizes, int n_in,
                              void* d_out, int out_size, void* d_ws, size_t ws_size,
                              hipStream_t stream) {
  (void)in_sizes; (void)n_in; (void)out_size; (void)ws_size;
  const float* x     = (const float*)d_in[0];
  const float* enc   = (const float*)d_in[1];
  // d_in[2] = self_mask (causal lower-triangular; applied analytically)
  const float* sa_wq = (const float*)d_in[3],  *sa_bq = (const float*)d_in[4];
  const float* sa_wk = (const float*)d_in[5],  *sa_bk = (const float*)d_in[6];
  const float* sa_wv = (const float*)d_in[7],  *sa_bv = (const float*)d_in[8];
  const float* sa_wo = (const float*)d_in[9],  *sa_bo = (const float*)d_in[10];
  const float* ca_wq = (const float*)d_in[11], *ca_bq = (const float*)d_in[12];
  const float* ca_wk = (const float*)d_in[13], *ca_bk = (const float*)d_in[14];
  const float* ca_wv = (const float*)d_in[15], *ca_bv = (const float*)d_in[16];
  const float* ca_wo = (const float*)d_in[17], *ca_bo = (const float*)d_in[18];
  const float* ff_w1 = (const float*)d_in[19], *ff_b1 = (const float*)d_in[20];
  const float* ff_w2 = (const float*)d_in[21], *ff_b2 = (const float*)d_in[22];
  const float* ln1g  = (const float*)d_in[23], *ln1b  = (const float*)d_in[24];
  const float* ln2g  = (const float*)d_in[25], *ln2b  = (const float*)d_in[26];
  const float* ln3g  = (const float*)d_in[27], *ln3b  = (const float*)d_in[28];

  char* ws = (char*)d_ws;
  size_t off = 0;
  auto alloc = [&](size_t bytes) -> char* {
    char* p = ws + off; off += (bytes + 255) & ~(size_t)255; return p;
  };
  u16* lnbuf = (u16*)alloc((size_t)NTOK * D_ * 2);
  u16* qbuf  = (u16*)alloc((size_t)NTOK * D_ * 2);
  u16* kbuf  = (u16*)alloc((size_t)NTOK * D_ * 2);
  u16* vbuf  = (u16*)alloc((size_t)NTOK * D_ * 2);
  u16* vtbuf = (u16*)alloc((size_t)NTOK * D_ * 2);   // per-head transposed V
  u16* abuf  = (u16*)alloc((size_t)NTOK * D_ * 2);   // attention output
  u16* encb  = (u16*)alloc((size_t)NTOK * D_ * 2);
  u16* ffh   = (u16*)alloc((size_t)NTOK * DFF_ * 2);
  u16* wqs = (u16*)alloc((size_t)D_ * D_ * 2);       // all weights stored [N][K]
  u16* wks = (u16*)alloc((size_t)D_ * D_ * 2);
  u16* wvs = (u16*)alloc((size_t)D_ * D_ * 2);
  u16* wos = (u16*)alloc((size_t)D_ * D_ * 2);
  u16* wqc = (u16*)alloc((size_t)D_ * D_ * 2);
  u16* wkc = (u16*)alloc((size_t)D_ * D_ * 2);
  u16* wvc = (u16*)alloc((size_t)D_ * D_ * 2);
  u16* woc = (u16*)alloc((size_t)D_ * D_ * 2);
  u16* w1t = (u16*)alloc((size_t)D_ * DFF_ * 2);
  u16* w2t = (u16*)alloc((size_t)DFF_ * D_ * 2);

  float* resid = (float*)d_out;   // residual stream lives in the output buffer
  (void)hipMemcpyAsync(resid, x, sizeof(float) * (size_t)NTOK * D_,
                       hipMemcpyDeviceToDevice, stream);

  const dim3 blk(256);
  const dim3 tblk(32, 8);
  auto tcvt = [&](const float* src, u16* dst, int K, int N) {
    transpose_cvt<<<dim3(N / 32, K / 32), tblk, 0, stream>>>(src, dst, K, N);
  };
  tcvt(sa_wq, wqs, D_, D_);  tcvt(sa_wk, wks, D_, D_);
  tcvt(sa_wv, wvs, D_, D_);  tcvt(sa_wo, wos, D_, D_);
  tcvt(ca_wq, wqc, D_, D_);  tcvt(ca_wk, wkc, D_, D_);
  tcvt(ca_wv, wvc, D_, D_);  tcvt(ca_wo, woc, D_, D_);
  tcvt(ff_w1, w1t, D_, DFF_); tcvt(ff_w2, w2t, DFF_, D_);
  cvt_f32_bf16<<<dim3(2048), blk, 0, stream>>>(enc, encb, NTOK * D_);

  auto gemm = [&](const u16* A, const u16* Wt, const float* bias,
                  u16* Cb, float* Cf, int M, int N, int K, int mode) {
    gemm_bf16_wmma<<<dim3(N / BN, M / BM), blk, 0, stream>>>(
        A, Wt, bias, Cb, Cf, M, N, K, mode);
  };
  const dim3 tvgrid(S_ / 32, DK_ / 32, B_ * H_);

  // ---- block 1: causal self-attention ----
  layernorm_bf16<<<dim3(NTOK), blk, 0, stream>>>(resid, ln1g, ln1b, lnbuf);
  gemm(lnbuf, wqs, sa_bq, qbuf, nullptr, NTOK, D_, D_, 0);
  gemm(lnbuf, wks, sa_bk, kbuf, nullptr, NTOK, D_, D_, 0);
  gemm(lnbuf, wvs, sa_bv, vbuf, nullptr, NTOK, D_, D_, 0);
  transpose_head<<<tvgrid, tblk, 0, stream>>>(vbuf, vtbuf);
  attn_wmma<<<dim3(S_ / 128, H_, B_), blk, 0, stream>>>(qbuf, kbuf, vtbuf, abuf, 1);
  gemm(abuf, wos, sa_bo, nullptr, resid, NTOK, D_, D_, 2);

  // ---- block 2: cross-attention over encoder output ----
  layernorm_bf16<<<dim3(NTOK), blk, 0, stream>>>(resid, ln2g, ln2b, lnbuf);
  gemm(lnbuf, wqc, ca_bq, qbuf, nullptr, NTOK, D_, D_, 0);
  gemm(encb,  wkc, ca_bk, kbuf, nullptr, NTOK, D_, D_, 0);
  gemm(encb,  wvc, ca_bv, vbuf, nullptr, NTOK, D_, D_, 0);
  transpose_head<<<tvgrid, tblk, 0, stream>>>(vbuf, vtbuf);
  attn_wmma<<<dim3(S_ / 128, H_, B_), blk, 0, stream>>>(qbuf, kbuf, vtbuf, abuf, 0);
  gemm(abuf, woc, ca_bo, nullptr, resid, NTOK, D_, D_, 2);

  // ---- block 3: FFN with exact GELU ----
  layernorm_bf16<<<dim3(NTOK), blk, 0, stream>>>(resid, ln3g, ln3b, lnbuf);
  gemm(lnbuf, w1t, ff_b1, ffh, nullptr, NTOK, DFF_, D_, 1);
  gemm(ffh,  w2t, ff_b2, nullptr, resid, NTOK, D_, DFF_, 2);
}
